// GraphiTLayer_54185307407181
// MI455X (gfx1250) — matrix operand order repaired and checked by reading, not compiled, not run on hardware
//
#include <hip/hip_runtime.h>

typedef __bf16 bf16_t;
typedef __attribute__((ext_vector_type(16))) __bf16 v16bf;
typedef __attribute__((ext_vector_type(8)))  __bf16 v8bf;
typedef __attribute__((ext_vector_type(8)))  float  v8f;
typedef int v4i_t __attribute__((vector_size(16)));

#define D_MODEL 256
#define N_HEAD  8
#define HD      32
#define D_FF    2048
#define BB      8
#define NN      1024
#define MTOT    (BB * NN)   // 8192 rows

// Epilogue modes
#define EP_NONE          0
#define EP_BIAS          1
#define EP_BIAS_RELU_BF  2
#define EP_ELU_RES       3
#define EP_BIAS_RES      4

// ---------------------------------------------------------------------------
// CDNA5 async global->LDS staging (ASYNCcnt path), with fallback
// ---------------------------------------------------------------------------
#if defined(__has_builtin)
#if __has_builtin(__builtin_amdgcn_global_load_async_to_lds_b128)
#define HAVE_ASYNC_LDS 1
#endif
#endif

#ifdef HAVE_ASYNC_LDS
#define ASYNC_LD128(lds_dst, glb_src)                                          \
  __builtin_amdgcn_global_load_async_to_lds_b128(                              \
      (__attribute__((address_space(1))) v4i_t*)(glb_src),                     \
      (__attribute__((address_space(3))) v4i_t*)(lds_dst), 0, 0)
static __device__ __forceinline__ void async_wait0() {
#if __has_builtin(__builtin_amdgcn_s_wait_asynccnt)
  __builtin_amdgcn_s_wait_asynccnt(0);
#else
  asm volatile("s_wait_asynccnt 0" ::: "memory");
#endif
}
#endif

// ---------------------------------------------------------------------------
// small converters
// ---------------------------------------------------------------------------
__global__ void cvt_bf16_kernel(const float* __restrict__ in, bf16_t* __restrict__ out, long n) {
  long i = (long)blockIdx.x * blockDim.x + threadIdx.x;
  if (i < n) out[i] = (bf16_t)in[i];
}

// out[c*R + r] = in[r*C + c]  (transpose + convert)
__global__ void cvt_bf16_t_kernel(const float* __restrict__ in, bf16_t* __restrict__ out, int R, int C) {
  long i = (long)blockIdx.x * blockDim.x + threadIdx.x;
  if (i < (long)R * C) {
    int r = (int)(i / C), c = (int)(i % C);
    out[(long)c * R + r] = (bf16_t)in[i];
  }
}

// ---------------------------------------------------------------------------
// Generic bf16 WMMA GEMM:  C[M,Nc] = epilogue( A[M,K] @ Bt[Nc,K]^T )
// Block = 256 threads = 8 waves (2m x 4n), block tile 64x256, wave tile 32x64
// -> 8 v_wmma per k-step per wave.  M % 64 == 0, Nc % 256 == 0, K % 32 == 0.
// ---------------------------------------------------------------------------
__global__ __launch_bounds__(256)
void gemm_bf16_kernel(const bf16_t* __restrict__ A, const bf16_t* __restrict__ Bt,
                      float* __restrict__ Cf, bf16_t* __restrict__ Cb,
                      const float* __restrict__ bias, const float* __restrict__ residual,
                      int M, int Nc, int K,
                      long strideA, long strideBt, long strideC, int ep)
{
  __shared__ __align__(16) bf16_t sA[64 * 32];
  __shared__ __align__(16) bf16_t sB[256 * 32];

  const int tid  = threadIdx.x;
  const int wave = tid >> 5;
  const int lane = tid & 31;
  const int half = lane >> 4;
  const int l16  = lane & 15;
  const int wm   = wave >> 2;        // 0..1
  const int wn   = wave & 3;         // 0..3

  A  += (long)blockIdx.z * strideA;
  Bt += (long)blockIdx.z * strideBt;
  const long cbase = (long)blockIdx.z * strideC;

  const int m0 = blockIdx.y * 64 + wm * 32;
  const int n0 = blockIdx.x * 256 + wn * 64;

  v8f acc[8] = {};   // [i m-subtile][j n-subtile] = acc[i*4+j]

  const int eA = tid * 8;                   // A staging: one 16B chunk
  const int rA = eA >> 5, cA = eA & 31;

  for (int k0 = 0; k0 < K; k0 += 32) {
    const bf16_t* ga = A + (long)(blockIdx.y * 64 + rA) * K + k0 + cA;
#ifdef HAVE_ASYNC_LDS
    ASYNC_LD128(sA + eA, ga);
#pragma unroll
    for (int j = 0; j < 4; ++j) {           // B staging: four 16B chunks
      const int e = tid * 32 + j * 8;
      const int r = e >> 5, c = e & 31;
      ASYNC_LD128(sB + e, Bt + (long)(blockIdx.x * 256 + r) * K + k0 + c);
    }
    if (k0 + 32 < K) __builtin_prefetch(ga + 32, 0, 1);
    async_wait0();
#else
    *(v8bf*)(sA + eA) = *(const v8bf*)ga;
#pragma unroll
    for (int j = 0; j < 4; ++j) {
      const int e = tid * 32 + j * 8;
      const int r = e >> 5, c = e & 31;
      *(v8bf*)(sB + e) = *(const v8bf*)(Bt + (long)(blockIdx.x * 256 + r) * K + k0 + c);
    }
    if (k0 + 32 < K) __builtin_prefetch(ga + 32, 0, 1);
#endif
    __syncthreads();

    // A fragments (16x32 each): lanes 0-15 -> K 0-15, lanes 16-31 -> K 16-31
    v16bf af[2];
#pragma unroll
    for (int i = 0; i < 2; ++i) {
      const int am = wm * 32 + i * 16 + l16;
      v8bf alo = *(const v8bf*)(sA + am * 32 + half * 16);
      v8bf ahi = *(const v8bf*)(sA + am * 32 + half * 16 + 8);
#pragma unroll
      for (int q = 0; q < 8; ++q) { af[i][q] = alo[q]; af[i][q + 8] = ahi[q]; }
    }
    // B fragments (32x16 each): lane n = l16, K = half*16 + i
    v16bf bfv[4];
#pragma unroll
    for (int j = 0; j < 4; ++j) {
      const int bn = wn * 64 + j * 16 + l16;
      bfv[j] = *(const v16bf*)(sB + bn * 32 + half * 16);
    }
#pragma unroll
    for (int i = 0; i < 2; ++i)
#pragma unroll
      for (int j = 0; j < 4; ++j)
        acc[i * 4 + j] = __builtin_amdgcn_wmma_f32_16x16x32_bf16(
            false, af[i], false, bfv[j], (short)0, acc[i * 4 + j], false, false);
    __syncthreads();
  }

  // ---- epilogue: C layout row = half*8+v, col = l16 ----
#pragma unroll
  for (int i = 0; i < 2; ++i) {
#pragma unroll
    for (int j = 0; j < 4; ++j) {
#pragma unroll
      for (int v = 0; v < 8; ++v) {
        const int row = m0 + i * 16 + half * 8 + v;
        const int col = n0 + j * 16 + l16;
        float x = acc[i * 4 + j][v];
        const long off = cbase + (long)row * Nc + col;
        if (ep == EP_BIAS) {
          Cf[off] = x + bias[col];
        } else if (ep == EP_BIAS_RELU_BF) {
          float y = x + bias[col];
          Cb[off] = (bf16_t)fmaxf(y, 0.0f);
        } else if (ep == EP_ELU_RES) {
          float e = (x > 0.0f) ? x : (__expf(x) - 1.0f);
          Cf[off] = residual[off] + e;
        } else if (ep == EP_BIAS_RES) {
          Cf[off] = residual[off] + x + bias[col];
        } else {
          Cf[off] = x;
        }
      }
    }
  }
}

// ---------------------------------------------------------------------------
// f1/f2 = Wh @ a1 / a2, plus build Wh^T in bf16.  grid = B*N, block = 256.
// ---------------------------------------------------------------------------
__global__ __launch_bounds__(256)
void f1f2_wht_kernel(const float* __restrict__ Wh, const float* __restrict__ a_gat,
                     float* __restrict__ f1, float* __restrict__ f2, bf16_t* __restrict__ WhT)
{
  __shared__ float r1[256], r2[256];
  const int row = blockIdx.x;                 // b*N + n
  const int b = row >> 10, n = row & (NN - 1);
  const int t = threadIdx.x;
  float w = Wh[(long)row * D_MODEL + t];
  WhT[((long)b * D_MODEL + t) * NN + n] = (bf16_t)w;
  r1[t] = w * a_gat[t];
  r2[t] = w * a_gat[D_MODEL + t];
  __syncthreads();
  for (int s = 128; s > 0; s >>= 1) {
    if (t < s) { r1[t] += r1[t + s]; r2[t] += r2[t + s]; }
    __syncthreads();
  }
  if (t == 0) { f1[row] = r1[0]; f2[row] = r2[0]; }
}

// ---------------------------------------------------------------------------
// GAT masked softmax row kernel -> attn in bf16.  grid = B*N, block = 256.
// ---------------------------------------------------------------------------
__global__ __launch_bounds__(256)
void gat_attn_kernel(const int* __restrict__ adj, const float* __restrict__ f1,
                     const float* __restrict__ f2, bf16_t* __restrict__ attn)
{
  __shared__ float red[256];
  const int row = blockIdx.x;                 // b*N + i
  const int b = row >> 10;
  const int t = threadIdx.x;
  const int* arow = adj + (long)row * NN;
  const float* f2b = f2 + (long)b * NN;
  const float fi = f1[row];

  float e[4]; float mx = -3.4e38f;
#pragma unroll
  for (int k = 0; k < 4; ++k) {
    int j = t + k * 256;
    float v = (arow[j] > 0) ? (fi + f2b[j]) : -9.0e15f;
    e[k] = v; mx = fmaxf(mx, v);
  }
  red[t] = mx; __syncthreads();
  for (int s = 128; s > 0; s >>= 1) { if (t < s) red[t] = fmaxf(red[t], red[t + s]); __syncthreads(); }
  mx = red[0]; __syncthreads();

  float sum = 0.0f;
#pragma unroll
  for (int k = 0; k < 4; ++k) { e[k] = __expf(e[k] - mx); sum += e[k]; }
  red[t] = sum; __syncthreads();
  for (int s = 128; s > 0; s >>= 1) { if (t < s) red[t] += red[t + s]; __syncthreads(); }
  const float inv = 1.0f / fmaxf(red[0], 1e-30f);

  bf16_t* orow = attn + (long)row * NN;
#pragma unroll
  for (int k = 0; k < 4; ++k) orow[t + k * 256] = (bf16_t)(e[k] * inv);
}

// ---------------------------------------------------------------------------
// LayerNorm over d=256; writes f32 and (optionally) bf16.  grid = rows.
// ---------------------------------------------------------------------------
__global__ __launch_bounds__(256)
void ln_kernel(const float* __restrict__ in, const float* __restrict__ g,
               const float* __restrict__ bb, float* __restrict__ outf,
               bf16_t* __restrict__ outb)
{
  __shared__ float red[256];
  const int row = blockIdx.x, t = threadIdx.x;
  const float x = in[(long)row * D_MODEL + t];
  red[t] = x; __syncthreads();
  for (int s = 128; s > 0; s >>= 1) { if (t < s) red[t] += red[t + s]; __syncthreads(); }
  const float mu = red[0] * (1.0f / D_MODEL);
  __syncthreads();
  const float xm = x - mu;
  red[t] = xm * xm; __syncthreads();
  for (int s = 128; s > 0; s >>= 1) { if (t < s) red[t] += red[t + s]; __syncthreads(); }
  const float r = rsqrtf(red[0] * (1.0f / D_MODEL) + 1e-5f);
  const float y = xm * r * g[t] + bb[t];
  outf[(long)row * D_MODEL + t] = y;
  if (outb) outb[(long)row * D_MODEL + t] = (bf16_t)y;
}

// ---------------------------------------------------------------------------
// qkv (f32, [B*N,768]) -> q,k as (B,H,N,32) bf16 and v^T as (B,H,32,N) bf16
// ---------------------------------------------------------------------------
__global__ void qkv_split_kernel(const float* __restrict__ qkv, bf16_t* __restrict__ q,
                                 bf16_t* __restrict__ k, bf16_t* __restrict__ vT)
{
  long i = (long)blockIdx.x * blockDim.x + threadIdx.x;  // over 8192*768
  const int c = (int)(i % (3 * D_MODEL));
  const long bn = i / (3 * D_MODEL);
  const int n = (int)(bn & (NN - 1));
  const int b = (int)(bn >> 10);
  const float v = qkv[i];
  const int which = c >> 8, cc = c & 255, h = cc >> 5, j = cc & 31;
  const long bh = (long)b * N_HEAD + h;
  if (which == 0)      q [(bh * NN + n) * HD + j] = (bf16_t)v;
  else if (which == 1) k [(bh * NN + n) * HD + j] = (bf16_t)v;
  else                 vT[(bh * HD + j) * NN + n] = (bf16_t)v;
}

// ---------------------------------------------------------------------------
// Flash attention: grid (N/64, B*H), block 128 = 4 waves, each wave 16 q-rows.
// Writes ctx directly in (B, N, H*32) bf16 layout for the out-proj GEMM.
// ---------------------------------------------------------------------------
__global__ __launch_bounds__(128)
void flash_attn_kernel(const bf16_t* __restrict__ q, const bf16_t* __restrict__ k,
                       const bf16_t* __restrict__ vT, bf16_t* __restrict__ ctx)
{
  __shared__ __align__(16) bf16_t sP[4][16 * 32];

  const int bh   = blockIdx.y;
  const int b    = bh / N_HEAD, h = bh % N_HEAD;
  const int wave = threadIdx.x >> 5;
  const int lane = threadIdx.x & 31;
  const int half = lane >> 4, l16 = lane & 15;
  const int q0   = blockIdx.x * 64 + wave * 16;

  const bf16_t* qb  = q  + (long)bh * NN * HD;
  const bf16_t* kb  = k  + (long)bh * NN * HD;
  const bf16_t* vtb = vT + (long)bh * HD * NN;
  bf16_t* P = sP[wave];

  // Q fragment (persistent)
  const int qm = q0 + l16;
  v8bf qlo = *(const v8bf*)(qb + (long)qm * HD + half * 16);
  v8bf qhi = *(const v8bf*)(qb + (long)qm * HD + half * 16 + 8);
  v16bf qf;
#pragma unroll
  for (int i = 0; i < 8; ++i) { qf[i] = qlo[i]; qf[i + 8] = qhi[i]; }

  v8f ctx0 = {}, ctx1 = {};
  float mst[8], lst[8];
#pragma unroll
  for (int v = 0; v < 8; ++v) { mst[v] = -3.4e38f; lst[v] = 0.0f; }
  const float scale = 0.17677669529663687f;   // 1/sqrt(32)

  for (int kk = 0; kk < NN; kk += 32) {
    // scores: 2 WMMAs (keys kk..kk+15, kk+16..kk+31)
    v16bf kf0 = *(const v16bf*)(kb + (long)(kk + l16) * HD + half * 16);
    v16bf kf1 = *(const v16bf*)(kb + (long)(kk + 16 + l16) * HD + half * 16);
    v8f s0 = {}, s1 = {};
    s0 = __builtin_amdgcn_wmma_f32_16x16x32_bf16(false, qf, false, kf0, (short)0, s0, false, false);
    s1 = __builtin_amdgcn_wmma_f32_16x16x32_bf16(false, qf, false, kf1, (short)0, s1, false, false);

    float pm[8];
#pragma unroll
    for (int v = 0; v < 8; ++v) {
      s0[v] *= scale; s1[v] *= scale;
      pm[v] = fmaxf(s0[v], s1[v]);
    }
#pragma unroll
    for (int msk = 1; msk < 16; msk <<= 1)
#pragma unroll
      for (int v = 0; v < 8; ++v) pm[v] = fmaxf(pm[v], __shfl_xor(pm[v], msk, 32));

    float alpha[8], rs[8];
#pragma unroll
    for (int v = 0; v < 8; ++v) {
      float mn = fmaxf(mst[v], pm[v]);
      alpha[v] = __expf(mst[v] - mn);
      mst[v] = mn;
      s0[v] = __expf(s0[v] - mn);
      s1[v] = __expf(s1[v] - mn);
      rs[v] = s0[v] + s1[v];
    }
#pragma unroll
    for (int msk = 1; msk < 16; msk <<= 1)
#pragma unroll
      for (int v = 0; v < 8; ++v) rs[v] += __shfl_xor(rs[v], msk, 32);
#pragma unroll
    for (int v = 0; v < 8; ++v) {
      lst[v] = lst[v] * alpha[v] + rs[v];
      ctx0[v] *= alpha[v];
      ctx1[v] *= alpha[v];
    }

    // P tile (16x32) -> per-wave LDS, then reload as A fragment
#pragma unroll
    for (int v = 0; v < 8; ++v) {
      P[(half * 8 + v) * 32 + l16]      = (bf16_t)s0[v];
      P[(half * 8 + v) * 32 + 16 + l16] = (bf16_t)s1[v];
    }
    asm volatile("s_wait_dscnt 0" ::: "memory");   // wave-private RAW on LDS

    v8bf plo = *(const v8bf*)(P + l16 * 32 + half * 16);
    v8bf phi = *(const v8bf*)(P + l16 * 32 + half * 16 + 8);
    v16bf pf;
#pragma unroll
    for (int i = 0; i < 8; ++i) { pf[i] = plo[i]; pf[i + 8] = phi[i]; }

    // V fragments from v^T: n = hd, K = key
    v16bf vf0 = *(const v16bf*)(vtb + (long)l16 * NN + kk + half * 16);
    v16bf vf1 = *(const v16bf*)(vtb + (long)(16 + l16) * NN + kk + half * 16);
    ctx0 = __builtin_amdgcn_wmma_f32_16x16x32_bf16(false, pf, false, vf0, (short)0, ctx0, false, false);
    ctx1 = __builtin_amdgcn_wmma_f32_16x16x32_bf16(false, pf, false, vf1, (short)0, ctx1, false, false);
  }

#pragma unroll
  for (int v = 0; v < 8; ++v) {
    const float inv = 1.0f / fmaxf(lst[v], 1e-30f);
    const int r = q0 + half * 8 + v;
    const long o = ((long)b * NN + r) * D_MODEL + h * HD;
    ctx[o + l16]      = (bf16_t)(ctx0[v] * inv);
    ctx[o + 16 + l16] = (bf16_t)(ctx1[v] * inv);
  }
}

// ---------------------------------------------------------------------------
// workspace layout
// ---------------------------------------------------------------------------
static constexpr size_t OF_WGATT  = 0;                                // 256x256 bf16
static constexpr size_t OF_INPROJ = OF_WGATT  + 256ul*256*2;          // 768x256 bf16
static constexpr size_t OF_OUTPRJ = OF_INPROJ + 768ul*256*2;
static constexpr size_t OF_LIN1   = OF_OUTPRJ + 256ul*256*2;          // 2048x256
static constexpr size_t OF_LIN2   = OF_LIN1   + 2048ul*256*2;         // 256x2048
static constexpr size_t OF_SRCBF  = OF_LIN2   + 256ul*2048*2;         // 8192x256 bf16
static constexpr size_t OF_WH     = OF_SRCBF  + (size_t)MTOT*256*2;   // 8192x256 f32
static constexpr size_t OF_WHT    = OF_WH     + (size_t)MTOT*256*4;   // 8x256x1024 bf16
static constexpr size_t OF_F1     = OF_WHT    + (size_t)MTOT*256*2;
static constexpr size_t OF_F2     = OF_F1     + (size_t)MTOT*4;
static constexpr size_t OF_ATTN   = OF_F2     + (size_t)MTOT*4;       // 8x1024x1024 bf16 (16MB)
static constexpr size_t OF_QKV    = OF_ATTN   + 8ul*1024*1024*2;      // 8192x768 f32 (24MB)
static constexpr size_t OF_FF1    = OF_ATTN;                          // reuse attn+qkv (32MB needed, 40MB avail)
static constexpr size_t OF_HP     = OF_QKV    + (size_t)MTOT*768*4;
static constexpr size_t OF_X      = OF_HP     + (size_t)MTOT*256*4;
static constexpr size_t OF_XBF    = OF_X      + (size_t)MTOT*256*4;
static constexpr size_t OF_QB     = OF_XBF    + (size_t)MTOT*256*2;
static constexpr size_t OF_KB     = OF_QB     + (size_t)MTOT*HD*2;
static constexpr size_t OF_VTB    = OF_KB     + (size_t)MTOT*HD*2;
static constexpr size_t OF_CTXBF  = OF_VTB    + (size_t)MTOT*HD*2;
static constexpr size_t OF_AO     = OF_CTXBF  + (size_t)MTOT*256*2;
static constexpr size_t OF_Y      = OF_AO     + (size_t)MTOT*256*4;
static constexpr size_t OF_YBF    = OF_Y      + (size_t)MTOT*256*4;
static constexpr size_t OF_Z      = OF_YBF    + (size_t)MTOT*256*2;

extern "C" void kernel_launch(void* const* d_in, const int* in_sizes, int n_in,
                              void* d_out, int out_size, void* d_ws, size_t ws_size,
                              hipStream_t stream) {
  (void)in_sizes; (void)n_in; (void)out_size; (void)ws_size;
  const float* src        = (const float*)d_in[0];
  const int*   adj        = (const int*)  d_in[1];
  const float* W_gat      = (const float*)d_in[2];
  const float* a_gat      = (const float*)d_in[3];
  const float* in_proj_w  = (const float*)d_in[4];
  const float* in_proj_b  = (const float*)d_in[5];
  const float* out_proj_w = (const float*)d_in[6];
  const float* out_proj_b = (const float*)d_in[7];
  const float* lin1_w     = (const float*)d_in[8];
  const float* lin1_b     = (const float*)d_in[9];
  const float* lin2_w     = (const float*)d_in[10];
  const float* lin2_b     = (const float*)d_in[11];
  const float* ln1_g = (const float*)d_in[12]; const float* ln1_b = (const float*)d_in[13];
  const float* ln2_g = (const float*)d_in[14]; const float* ln2_b = (const float*)d_in[15];
  const float* ln3_g = (const float*)d_in[16]; const float* ln3_b = (const float*)d_in[17];

  char* ws = (char*)d_ws;
  bf16_t* wgatT  = (bf16_t*)(ws + OF_WGATT);
  bf16_t* inprj  = (bf16_t*)(ws + OF_INPROJ);
  bf16_t* outprj = (bf16_t*)(ws + OF_OUTPRJ);
  bf16_t* l1bf   = (bf16_t*)(ws + OF_LIN1);
  bf16_t* l2bf   = (bf16_t*)(ws + OF_LIN2);
  bf16_t* srcbf  = (bf16_t*)(ws + OF_SRCBF);
  float*  Wh     = (float*) (ws + OF_WH);
  bf16_t* WhT    = (bf16_t*)(ws + OF_WHT);
  float*  f1     = (float*) (ws + OF_F1);
  float*  f2     = (float*) (ws + OF_F2);
  bf16_t* attn   = (bf16_t*)(ws + OF_ATTN);
  float*  qkv    = (float*) (ws + OF_QKV);
  bf16_t* ff1    = (bf16_t*)(ws + OF_FF1);
  float*  hp     = (float*) (ws + OF_HP);
  float*  x      = (float*) (ws + OF_X);
  bf16_t* xbf    = (bf16_t*)(ws + OF_XBF);
  bf16_t* qb     = (bf16_t*)(ws + OF_QB);
  bf16_t* kb     = (bf16_t*)(ws + OF_KB);
  bf16_t* vtb    = (bf16_t*)(ws + OF_VTB);
  bf16_t* ctxbf  = (bf16_t*)(ws + OF_CTXBF);
  float*  ao     = (float*) (ws + OF_AO);
  float*  y      = (float*) (ws + OF_Y);
  bf16_t* ybf    = (bf16_t*)(ws + OF_YBF);
  float*  z      = (float*) (ws + OF_Z);
  float*  outp   = (float*) d_out;

  auto cdiv = [](long a, long b) { return (int)((a + b - 1) / b); };

  // 0) weight / input conversion
  cvt_bf16_t_kernel<<<cdiv(256*256,256),256,0,stream>>>(W_gat, wgatT, 256, 256);
  cvt_bf16_kernel<<<cdiv(768*256,256),256,0,stream>>>(in_proj_w,  inprj,  768*256);
  cvt_bf16_kernel<<<cdiv(256*256,256),256,0,stream>>>(out_proj_w, outprj, 256*256);
  cvt_bf16_kernel<<<cdiv(2048*256,256),256,0,stream>>>(lin1_w, l1bf, 2048*256);
  cvt_bf16_kernel<<<cdiv(256*2048,256),256,0,stream>>>(lin2_w, l2bf, 256*2048);
  cvt_bf16_kernel<<<cdiv((long)MTOT*256,256),256,0,stream>>>(src, srcbf, (long)MTOT*256);

  // 1) Wh = src @ W_gat
  gemm_bf16_kernel<<<dim3(256/256, MTOT/64, 1),256,0,stream>>>(
      srcbf, wgatT, Wh, nullptr, nullptr, nullptr, MTOT, 256, 256, 0, 0, 0, EP_NONE);

  // 2) f1, f2, Wh^T
  f1f2_wht_kernel<<<MTOT,256,0,stream>>>(Wh, a_gat, f1, f2, WhT);

  // 3) GAT masked softmax -> attn (bf16)
  gat_attn_kernel<<<MTOT,256,0,stream>>>(adj, f1, f2, attn);

  // 4) hp = src + elu(attn @ Wh)   (batched over B)
  gemm_bf16_kernel<<<dim3(256/256, NN/64, BB),256,0,stream>>>(
      attn, WhT, hp, nullptr, nullptr, src, NN, 256, NN,
      (long)NN*NN, (long)256*NN, (long)NN*256, EP_ELU_RES);

  // 5) x = LN1(hp)
  ln_kernel<<<MTOT,256,0,stream>>>(hp, ln1_g, ln1_b, x, xbf);

  // 6) qkv = x @ in_proj_w^T + b
  gemm_bf16_kernel<<<dim3(768/256, MTOT/64, 1),256,0,stream>>>(
      xbf, inprj, qkv, nullptr, in_proj_b, nullptr, MTOT, 768, 256, 0, 0, 0, EP_BIAS);

  // 7) split into q, k, v^T
  qkv_split_kernel<<<cdiv((long)MTOT*768,256),256,0,stream>>>(qkv, qb, kb, vtb);

  // 8) flash attention -> ctx (bf16, (B,N,256))
  flash_attn_kernel<<<dim3(NN/64, BB*N_HEAD),128,0,stream>>>(qb, kb, vtb, ctxbf);

  // 9) ao = x + ctx @ out_proj_w^T + b
  gemm_bf16_kernel<<<dim3(256/256, MTOT/64, 1),256,0,stream>>>(
      ctxbf, outprj, ao, nullptr, out_proj_b, x, MTOT, 256, 256, 0, 0, 0, EP_BIAS_RES);

  // 10) y = LN2(ao)
  ln_kernel<<<MTOT,256,0,stream>>>(ao, ln2_g, ln2_b, y, ybf);

  // 11) ff1 = relu(y @ lin1_w^T + b)  (bf16 out)
  gemm_bf16_kernel<<<dim3(D_FF/256, MTOT/64, 1),256,0,stream>>>(
      ybf, l1bf, nullptr, ff1, lin1_b, nullptr, MTOT, D_FF, 256, 0, 0, 0, EP_BIAS_RELU_BF);

  // 12) z = y + ff1 @ lin2_w^T + b
  gemm_bf16_kernel<<<dim3(256/256, MTOT/64, 1),256,0,stream>>>(
      ff1, l2bf, z, nullptr, lin2_b, y, MTOT, 256, D_FF, 0, 0, 0, EP_BIAS_RES);

  // 13) out = LN3(z)
  ln_kernel<<<MTOT,256,0,stream>>>(z, ln3_g, ln3_b, outp, nullptr);
}
